// EquivariantModel_62551903699067
// MI455X (gfx1250) — matrix operand order repaired and checked by reading, not compiled, not run on hardware
//
#include <hip/hip_runtime.h>

// ---------------------------------------------------------------------------
// Types for CDNA5 WMMA (gfx1250, wave32)
// ---------------------------------------------------------------------------
typedef __attribute__((ext_vector_type(16))) __bf16 v16bf;
typedef __attribute__((ext_vector_type(8)))  __bf16 v8bf;
typedef __attribute__((ext_vector_type(8)))  float  v8f;
typedef __attribute__((ext_vector_type(2)))  float  v2f;

#define TILE_B    32
#define LDSTRIDE  136                       // 128 + 8 pad (bank skew), 16B aligned rows
#define PLANE     (TILE_B * LDSTRIDE)       // one i-plane of a v-buffer
#define SBUF      (TILE_B * LDSTRIDE)       // one s-buffer
#define LDS_FLOATS (3*SBUF + 3*3*SBUF + SBUF)
#define LDS_BYTES  (LDS_FLOATS * 4)         // 226304 B  (< 320 KB WGP LDS)

#define TANH_GAIN 1.5927116870880127f

// ---------------------------------------------------------------------------
// Weight table: order matches setup_inputs() (d_in[1+i]); transposed to [N][K]
// bf16 in d_ws with scale constants folded in.
// ---------------------------------------------------------------------------
enum WId { B1_L1_W0=0, B1_L1_W1, B1_L2_W0, B1_L2_W1,
           B1_TP_SS, B1_TP_VV, B1_TP_SV, B1_TP_VS,
           B1_G_WS, B1_G_WG, B1_G_WV, B1_O_W0, B1_O_W1,
           B2_L1_W0, B2_L1_W1, B2_L2_W0, B2_L2_W1,
           B2_TP_SS, B2_TP_VV, B2_TP_SV, B2_TP_VS,
           B2_G_WS, B2_G_WG, B2_G_WV, B2_O_W0, B2_O_W1,
           F_W0, F_W1, NWT };

constexpr int WKd[NWT] = {64,64,64,64, 4096,4096,4096,4096, 128,128,128, 128,128,
                          128,128,128,128, 16384,16384,16384,16384, 128,128,128, 128,128,
                          128,128};
constexpr int WNd[NWT] = {64,64,64,64, 128,128,128,128, 128,128,128, 128,128,
                          128,128,128,128, 128,128,128,128, 128,128,128, 128,128,
                          64,64};

struct WOffs { size_t v[NWT + 1]; };
constexpr WOffs make_offs() {
  WOffs o{}; size_t a = 0;
  for (int j = 0; j < NWT; ++j) { o.v[j] = a; a += (size_t)WKd[j] * WNd[j]; }
  o.v[NWT] = a; return o;
}
constexpr WOffs WOFF = make_offs();

// scale constants folded into weights at prep time
#define SC64   0.125f                  // 1/sqrt(64)
#define SC128  0.08838834764831845f    // 1/sqrt(128)
#define C1TP   0.011048543456039806f   // 1/(64*sqrt(2))
#define C1TPV  0.006378864105964106f   // C1TP/sqrt(3)
#define C2TP   0.005524271728019903f   // 1/(128*sqrt(2))
#define C2TPV  0.003189432052982053f   // C2TP/sqrt(3)
constexpr float WSCALE[NWT] = {
  SC64,SC64,SC64,SC64,  C1TP,C1TPV,C1TP,C1TP,  SC128,SC128,SC128, SC128,SC128,
  SC128,SC128,SC128,SC128,  C2TP,C2TPV,C2TP,C2TP,  SC128,SC128,SC128, SC128,SC128,
  SC128,SC128 };

// ---------------------------------------------------------------------------
// Weight prep: f32 (K rows x N cols) -> bf16 transposed [N][K], scaled.
// ---------------------------------------------------------------------------
__global__ void prep_w(const float* __restrict__ src, __bf16* __restrict__ dst,
                       int K, int N, float scale) {
  int idx = blockIdx.x * 256 + threadIdx.x;
  if (idx >= K * N) return;
  int n = idx / K, k = idx - n * K;
  dst[idx] = (__bf16)(src[(size_t)k * N + n] * scale);
}

// ---------------------------------------------------------------------------
// Fragment helpers (wave32 layouts per CDNA5 ISA 7.12.2)
//   A (16x32 bf16): lane = row (lane&15); khalf=lane>>4 -> K {0-7,16-23} / {8-15,24-31}
//   B (32x16 bf16): lane = col (lane&15); khalf -> K {0-7,16-23} / {8-15,24-31}
//   C/D (16x16 f32): lane = col (lane&15); rows = (lane>>4)*8 + j
// ---------------------------------------------------------------------------
__device__ __forceinline__ v8f wmma_bf16(v16bf a, v16bf b, v8f c) {
  return __builtin_amdgcn_wmma_f32_16x16x32_bf16(false, a, false, b, (short)0, c,
                                                 false, false);
}

__device__ __forceinline__ v16bf fragB(const __bf16* Brow, int kc, int khalf) {
  const __bf16* p = Brow + kc + khalf * 8;
  v8bf lo = *reinterpret_cast<const v8bf*>(p);
  v8bf hi = *reinterpret_cast<const v8bf*>(p + 16);
  return __builtin_shufflevector(lo, hi, 0,1,2,3,4,5,6,7,8,9,10,11,12,13,14,15);
}

__device__ __forceinline__ v16bf fragA_f32(const float* row, int kc, int khalf) {
  const float* p = row + kc + khalf * 8;
  float4 x0 = *(const float4*)(p);
  float4 x1 = *(const float4*)(p + 4);
  float4 y0 = *(const float4*)(p + 16);
  float4 y1 = *(const float4*)(p + 20);
  v16bf a;
  a[0]=(__bf16)x0.x; a[1]=(__bf16)x0.y; a[2]=(__bf16)x0.z; a[3]=(__bf16)x0.w;
  a[4]=(__bf16)x1.x; a[5]=(__bf16)x1.y; a[6]=(__bf16)x1.z; a[7]=(__bf16)x1.w;
  a[8]=(__bf16)y0.x; a[9]=(__bf16)y0.y; a[10]=(__bf16)y0.z; a[11]=(__bf16)y0.w;
  a[12]=(__bf16)y1.x; a[13]=(__bf16)y1.y; a[14]=(__bf16)y1.z; a[15]=(__bf16)y1.w;
  return a;
}

// A[row, k] = s * rowv[v0 + klocal]  (outer product slice; packed f32 math)
__device__ __forceinline__ v16bf frag_outer(float s, const float* rowv, int v0, int khalf) {
  const v2f* p = reinterpret_cast<const v2f*>(rowv + v0 + khalf * 8);
  v2f sv = {s, s};
  v16bf a;
  #pragma unroll
  for (int j = 0; j < 4; ++j) {
    v2f lo = p[j] * sv;        // v_pk_mul_f32
    v2f hi = p[j + 8] * sv;    // +16 floats
    a[2*j]     = (__bf16)lo.x;  a[2*j + 1]     = (__bf16)lo.y;
    a[2*j + 8] = (__bf16)hi.x;  a[2*j + 9]     = (__bf16)hi.y;
  }
  return a;
}

// A[row, k] = sum_i v1[i][row][u] * v2[i][row][v0 + klocal]  (packed fma)
__device__ __forceinline__ v16bf frag_vv(const float* v1r, const float* v2r,
                                         int u, int v0, int khalf) {
  v2f a0 = {v1r[u], v1r[u]};
  v2f a1 = {v1r[PLANE + u], v1r[PLANE + u]};
  v2f a2 = {v1r[2 * PLANE + u], v1r[2 * PLANE + u]};
  const v2f* p0 = reinterpret_cast<const v2f*>(v2r + v0 + khalf * 8);
  const v2f* p1 = reinterpret_cast<const v2f*>(v2r + PLANE + v0 + khalf * 8);
  const v2f* p2 = reinterpret_cast<const v2f*>(v2r + 2 * PLANE + v0 + khalf * 8);
  v16bf a;
  #pragma unroll
  for (int j = 0; j < 4; ++j) {
    v2f lo = p0[j] * a0 + p1[j] * a1 + p2[j] * a2;          // pk_mul + 2x pk_fma
    v2f hi = p0[j + 8] * a0 + p1[j + 8] * a1 + p2[j + 8] * a2;
    a[2*j]     = (__bf16)lo.x;  a[2*j + 1] = (__bf16)lo.y;
    a[2*j + 8] = (__bf16)hi.x;  a[2*j + 9] = (__bf16)hi.y;
  }
  return a;
}

__device__ __forceinline__ void store_acc(float* C, const v8f& a0, const v8f& a1,
                                          int n0, int lane) {
  int col = n0 + (lane & 15);
  int r0  = (lane >> 4) * 8;
  #pragma unroll
  for (int j = 0; j < 8; ++j) {
    C[(r0 + j) * LDSTRIDE + col]      = a0[j];
    C[(r0 + j + 16) * LDSTRIDE + col] = a1[j];
  }
}

#define VZERO8 {0.f,0.f,0.f,0.f,0.f,0.f,0.f,0.f}

// paired GEMM: computes N-tiles n0 and n0+16 for 32 batch rows.
// C = A[32 x K] (f32 LDS) @ Bt^T  (Bt bf16 [N][K])
__device__ __forceinline__ void gemm_lin2(const float* A, const __bf16* Bt, int K,
                                          float* C, int n0, int lane) {
  int row = lane & 15, khalf = lane >> 4;
  v8f acc00 = VZERO8, acc01 = VZERO8, acc10 = VZERO8, acc11 = VZERO8;
  const __bf16* Brow0 = Bt + (size_t)(n0 + row) * K;
  const __bf16* Brow1 = Brow0 + (size_t)16 * K;
  const float* Ar0 = A + row * LDSTRIDE;
  const float* Ar1 = Ar0 + 16 * LDSTRIDE;
  for (int kc = 0; kc < K; kc += 32) {
    v16bf b0 = fragB(Brow0, kc, khalf);
    v16bf b1 = fragB(Brow1, kc, khalf);
    v16bf a0 = fragA_f32(Ar0, kc, khalf);
    v16bf a1 = fragA_f32(Ar1, kc, khalf);
    acc00 = wmma_bf16(a0, b0, acc00);
    acc10 = wmma_bf16(a1, b0, acc10);
    acc01 = wmma_bf16(a0, b1, acc01);
    acc11 = wmma_bf16(a1, b1, acc11);
  }
  store_acc(C, acc00, acc10, n0, lane);
  store_acc(C, acc01, acc11, n0 + 16, lane);
}

// tensor-product scalar channel (paired N-tiles):
// Sout[:, n0..n0+31] = (s1⊗s2)@Wss + (Σ_i v1_i⊗v2_i)@Wvv
template<int DIN>
__device__ __forceinline__ void tp_s2(const float* S1, const float* S2,
                                      const float* V1, const float* V2,
                                      const __bf16* WssT, const __bf16* WvvT,
                                      float* Sout, int n0, int lane) {
  constexpr int KTP = DIN * DIN;
  int row = lane & 15, khalf = lane >> 4;
  v8f acc00 = VZERO8, acc01 = VZERO8, acc10 = VZERO8, acc11 = VZERO8;
  const __bf16* Bss0 = WssT + (size_t)(n0 + row) * KTP;
  const __bf16* Bss1 = Bss0 + (size_t)16 * KTP;
  const __bf16* Bvv0 = WvvT + (size_t)(n0 + row) * KTP;
  const __bf16* Bvv1 = Bvv0 + (size_t)16 * KTP;
  const float* s1r0 = S1 + row * LDSTRIDE; const float* s1r1 = s1r0 + 16 * LDSTRIDE;
  const float* s2r0 = S2 + row * LDSTRIDE; const float* s2r1 = s2r0 + 16 * LDSTRIDE;
  const float* v1r0 = V1 + row * LDSTRIDE; const float* v1r1 = v1r0 + 16 * LDSTRIDE;
  const float* v2r0 = V2 + row * LDSTRIDE; const float* v2r1 = v2r0 + 16 * LDSTRIDE;
  #pragma unroll 2
  for (int kc = 0; kc < KTP; kc += 32) {
    int u = kc / DIN, v0 = kc % DIN;
    __builtin_prefetch(Bss0 + kc + 1024, 0, 1);
    __builtin_prefetch(Bss1 + kc + 1024, 0, 1);
    v16bf b0 = fragB(Bss0, kc, khalf);
    v16bf b1 = fragB(Bss1, kc, khalf);
    v16bf a0 = frag_outer(s1r0[u], s2r0, v0, khalf);
    v16bf a1 = frag_outer(s1r1[u], s2r1, v0, khalf);
    acc00 = wmma_bf16(a0, b0, acc00);
    acc10 = wmma_bf16(a1, b0, acc10);
    acc01 = wmma_bf16(a0, b1, acc01);
    acc11 = wmma_bf16(a1, b1, acc11);
  }
  #pragma unroll 2
  for (int kc = 0; kc < KTP; kc += 32) {
    int u = kc / DIN, v0 = kc % DIN;
    __builtin_prefetch(Bvv0 + kc + 1024, 0, 1);
    __builtin_prefetch(Bvv1 + kc + 1024, 0, 1);
    v16bf b0 = fragB(Bvv0, kc, khalf);
    v16bf b1 = fragB(Bvv1, kc, khalf);
    v16bf a0 = frag_vv(v1r0, v2r0, u, v0, khalf);
    v16bf a1 = frag_vv(v1r1, v2r1, u, v0, khalf);
    acc00 = wmma_bf16(a0, b0, acc00);
    acc10 = wmma_bf16(a1, b0, acc10);
    acc01 = wmma_bf16(a0, b1, acc01);
    acc11 = wmma_bf16(a1, b1, acc11);
  }
  store_acc(Sout, acc00, acc10, n0, lane);
  store_acc(Sout, acc01, acc11, n0 + 16, lane);
}

// tensor-product vector channel (one i, paired N-tiles):
// Vout_i[:, n0..n0+31] = (s1⊗v2_i)@Wsv + (v1_i⊗s2)@Wvs
template<int DIN>
__device__ __forceinline__ void tp_v2(const float* S1, const float* S2,
                                      const float* V1, const float* V2,
                                      const __bf16* WsvT, const __bf16* WvsT,
                                      float* Vout_i, int i, int n0, int lane) {
  constexpr int KTP = DIN * DIN;
  int row = lane & 15, khalf = lane >> 4;
  v8f acc00 = VZERO8, acc01 = VZERO8, acc10 = VZERO8, acc11 = VZERO8;
  const __bf16* Bsv0 = WsvT + (size_t)(n0 + row) * KTP;
  const __bf16* Bsv1 = Bsv0 + (size_t)16 * KTP;
  const __bf16* Bvs0 = WvsT + (size_t)(n0 + row) * KTP;
  const __bf16* Bvs1 = Bvs0 + (size_t)16 * KTP;
  const float* s1r0 = S1 + row * LDSTRIDE;             const float* s1r1 = s1r0 + 16 * LDSTRIDE;
  const float* s2r0 = S2 + row * LDSTRIDE;             const float* s2r1 = s2r0 + 16 * LDSTRIDE;
  const float* v1r0 = V1 + i * PLANE + row * LDSTRIDE; const float* v1r1 = v1r0 + 16 * LDSTRIDE;
  const float* v2r0 = V2 + i * PLANE + row * LDSTRIDE; const float* v2r1 = v2r0 + 16 * LDSTRIDE;
  #pragma unroll 2
  for (int kc = 0; kc < KTP; kc += 32) {            // s1 ⊗ v2_i
    int u = kc / DIN, v0 = kc % DIN;
    __builtin_prefetch(Bsv0 + kc + 1024, 0, 1);
    __builtin_prefetch(Bsv1 + kc + 1024, 0, 1);
    v16bf b0 = fragB(Bsv0, kc, khalf);
    v16bf b1 = fragB(Bsv1, kc, khalf);
    v16bf a0 = frag_outer(s1r0[u], v2r0, v0, khalf);
    v16bf a1 = frag_outer(s1r1[u], v2r1, v0, khalf);
    acc00 = wmma_bf16(a0, b0, acc00);
    acc10 = wmma_bf16(a1, b0, acc10);
    acc01 = wmma_bf16(a0, b1, acc01);
    acc11 = wmma_bf16(a1, b1, acc11);
  }
  #pragma unroll 2
  for (int kc = 0; kc < KTP; kc += 32) {            // v1_i ⊗ s2
    int u = kc / DIN, v0 = kc % DIN;
    __builtin_prefetch(Bvs0 + kc + 1024, 0, 1);
    __builtin_prefetch(Bvs1 + kc + 1024, 0, 1);
    v16bf b0 = fragB(Bvs0, kc, khalf);
    v16bf b1 = fragB(Bvs1, kc, khalf);
    v16bf a0 = frag_outer(v1r0[u], s2r0, v0, khalf);
    v16bf a1 = frag_outer(v1r1[u], s2r1, v0, khalf);
    acc00 = wmma_bf16(a0, b0, acc00);
    acc10 = wmma_bf16(a1, b0, acc10);
    acc01 = wmma_bf16(a0, b1, acc01);
    acc11 = wmma_bf16(a1, b1, acc11);
  }
  store_acc(Vout_i, acc00, acc10, n0, lane);
  store_acc(Vout_i, acc01, acc11, n0 + 16, lane);
}

// ---------------------------------------------------------------------------
// One equivariant block.  In/out activations live in Sin/Vin; S1,S2,V1,V2,G
// are scratch.  DIN = input feature dim (64 block1, 128 block2); tp/gate/out
// are always 128 wide.
// ---------------------------------------------------------------------------
template<int DIN>
__device__ void run_block(float* Sin, float* Vin,
                          float* S1, float* S2, float* V1, float* V2, float* G,
                          const __bf16* W, int wb, int lane, int wave, int tid) {
  constexpr int NPAIR = DIN / 32;    // paired N-tiles for lin outputs
  const __bf16* w_l1_s = W + WOFF.v[wb + 0];
  const __bf16* w_l1_v = W + WOFF.v[wb + 1];
  const __bf16* w_l2_s = W + WOFF.v[wb + 2];
  const __bf16* w_l2_v = W + WOFF.v[wb + 3];
  const __bf16* w_ss   = W + WOFF.v[wb + 4];
  const __bf16* w_vv   = W + WOFF.v[wb + 5];
  const __bf16* w_sv   = W + WOFF.v[wb + 6];
  const __bf16* w_vs   = W + WOFF.v[wb + 7];
  const __bf16* w_gs   = W + WOFF.v[wb + 8];
  const __bf16* w_gg   = W + WOFF.v[wb + 9];
  const __bf16* w_gv   = W + WOFF.v[wb + 10];
  const __bf16* w_o0   = W + WOFF.v[wb + 11];
  const __bf16* w_o1   = W + WOFF.v[wb + 12];

  // ---- lin1 & lin2 (K=DIN, N=DIN): Sin,Vin -> S1,S2,V1,V2
  for (int t = wave; t < 8 * NPAIR; t += 8) {
    int g = t / NPAIR, n0 = (t % NPAIR) * 32;
    const float* A; const __bf16* Bt; float* C;
    if (g == 0)      { A = Sin;                  Bt = w_l1_s; C = S1; }
    else if (g == 1) { A = Sin;                  Bt = w_l2_s; C = S2; }
    else if (g < 5)  { A = Vin + (g - 2) * PLANE; Bt = w_l1_v; C = V1 + (g - 2) * PLANE; }
    else             { A = Vin + (g - 5) * PLANE; Bt = w_l2_v; C = V2 + (g - 5) * PLANE; }
    gemm_lin2(A, Bt, DIN, C, n0, lane);
  }
  __syncthreads();

  // ---- tensor product (K = DIN*DIN, N=128): -> Sin (s), Vin (v)
  // 4 s-pairs + 12 v-pairs = 16 tasks, 2 per wave
  for (int t = wave; t < 16; t += 8) {
    if (t < 4) {
      tp_s2<DIN>(S1, S2, V1, V2, w_ss, w_vv, Sin, t * 32, lane);
    } else {
      int idx = t - 4, i = idx >> 2, n0 = (idx & 3) * 32;
      tp_v2<DIN>(S1, S2, V1, V2, w_sv, w_vs, Vin + i * PLANE, i, n0, lane);
    }
  }
  __syncthreads();

  // ---- gate GEMMs (K=128, N=128): sl->S1, g->G, vl->V1  (20 pair-tasks)
  for (int t = wave; t < 20; t += 8) {
    int g = t >> 2, n0 = (t & 3) * 32;
    const float* A; const __bf16* Bt; float* C;
    if (g == 0)      { A = Sin; Bt = w_gs; C = S1; }
    else if (g == 1) { A = Sin; Bt = w_gg; C = G;  }
    else             { A = Vin + (g - 2) * PLANE; Bt = w_gv; C = V1 + (g - 2) * PLANE; }
    gemm_lin2(A, Bt, 128, C, n0, lane);
  }
  __syncthreads();

  // ---- gate nonlinearity (elementwise, all 256 threads)
  for (int idx = tid; idx < TILE_B * 128; idx += 256) {
    int b = idx >> 7, u = idx & 127;
    int o = b * LDSTRIDE + u;
    S1[o] = TANH_GAIN * tanhf(S1[o]);
    float gg = TANH_GAIN * tanhf(G[o]);
    V1[o]             *= gg;
    V1[o + PLANE]     *= gg;
    V1[o + 2 * PLANE] *= gg;
  }
  __syncthreads();

  // ---- output linear (K=128, N=128): S1,V1 -> Sin,Vin  (16 pair-tasks)
  for (int t = wave; t < 16; t += 8) {
    int g = t >> 2, n0 = (t & 3) * 32;
    if (g == 0) gemm_lin2(S1, w_o0, 128, Sin, n0, lane);
    else        gemm_lin2(V1 + (g - 1) * PLANE, w_o1, 128, Vin + (g - 1) * PLANE, n0, lane);
  }
  __syncthreads();
}

// ---------------------------------------------------------------------------
// Main fused kernel: one 32-row batch tile per workgroup, 8 wave32s.
// ---------------------------------------------------------------------------
__global__ __launch_bounds__(256)
void equi_main(const float* __restrict__ x, const __bf16* __restrict__ W,
               float* __restrict__ out) {
  extern __shared__ float lds[];
  float* S0 = lds;
  float* S1 = S0 + SBUF;
  float* S2 = S1 + SBUF;
  float* V0 = S2 + SBUF;
  float* V1 = V0 + 3 * SBUF;
  float* V2 = V1 + 3 * SBUF;
  float* G  = V2 + 3 * SBUF;

  int tid  = threadIdx.x;
  int lane = tid & 31;
  int wave = __builtin_amdgcn_readfirstlane(tid >> 5);   // provably wave-uniform
  int b0 = blockIdx.x * TILE_B;

  // stage input: s = x[:, :64]; v[i][b][u] = x[b, 64 + u*3 + i]
  for (int idx = tid; idx < TILE_B * 256; idx += 256) {
    int b = idx >> 8, c = idx & 255;
    float val = x[(size_t)(b0 + b) * 256 + c];
    if (c < 64) S0[b * LDSTRIDE + c] = val;
    else {
      int cc = c - 64, u = cc / 3, i = cc - 3 * u;
      V0[i * PLANE + b * LDSTRIDE + u] = val;
    }
  }
  __syncthreads();

  run_block<64>(S0, V0, S1, S2, V1, V2, G, W, B1_L1_W0, lane, wave, tid);
  run_block<128>(S0, V0, S1, S2, V1, V2, G, W, B2_L1_W0, lane, wave, tid);

  // final linear (K=128, N=64) straight to global f32: 2 s-pairs + 6 v-pairs
  const __bf16* fw0 = W + WOFF.v[F_W0];
  const __bf16* fw1 = W + WOFF.v[F_W1];
  int row = lane & 15, khalf = lane >> 4;
  {
    int t = wave;  // exactly one pair-task per wave
    v8f acc00 = VZERO8, acc01 = VZERO8, acc10 = VZERO8, acc11 = VZERO8;
    const float* A; const __bf16* Bt; int n0, i = 0; bool isv;
    if (t < 2) { n0 = t * 32; A = S0; Bt = fw0; isv = false; }
    else { int idx = t - 2; i = idx >> 1; n0 = (idx & 1) * 32; A = V0 + i * PLANE; Bt = fw1; isv = true; }
    const __bf16* Brow0 = Bt + (size_t)(n0 + row) * 128;
    const __bf16* Brow1 = Brow0 + (size_t)16 * 128;
    const float* Ar0 = A + row * LDSTRIDE;
    const float* Ar1 = Ar0 + 16 * LDSTRIDE;
    for (int kc = 0; kc < 128; kc += 32) {
      v16bf b0 = fragB(Brow0, kc, khalf);
      v16bf b1 = fragB(Brow1, kc, khalf);
      v16bf a0 = fragA_f32(Ar0, kc, khalf);
      v16bf a1 = fragA_f32(Ar1, kc, khalf);
      acc00 = wmma_bf16(a0, b0, acc00);
      acc10 = wmma_bf16(a1, b0, acc10);
      acc01 = wmma_bf16(a0, b1, acc01);
      acc11 = wmma_bf16(a1, b1, acc11);
    }
    int col0 = n0 + (lane & 15);
    int col1 = col0 + 16;
    int r0   = khalf * 8;
    #pragma unroll
    for (int j = 0; j < 8; ++j) {
      size_t br0 = (size_t)(b0 + r0 + j) * 256;
      size_t br1 = (size_t)(b0 + r0 + 16 + j) * 256;
      if (!isv) {
        out[br0 + col0] = acc00[j];
        out[br1 + col0] = acc10[j];
        out[br0 + col1] = acc01[j];
        out[br1 + col1] = acc11[j];
      } else {
        out[br0 + 64 + col0 * 3 + i] = acc00[j];
        out[br1 + 64 + col0 * 3 + i] = acc10[j];
        out[br0 + 64 + col1 * 3 + i] = acc01[j];
        out[br1 + 64 + col1 * 3 + i] = acc11[j];
      }
    }
  }
}

// ---------------------------------------------------------------------------
// Host entry
// ---------------------------------------------------------------------------
extern "C" void kernel_launch(void* const* d_in, const int* in_sizes, int n_in,
                              void* d_out, int out_size, void* d_ws, size_t ws_size,
                              hipStream_t stream) {
  (void)in_sizes; (void)n_in; (void)out_size; (void)ws_size;
  const float* x = (const float*)d_in[0];
  __bf16* W = reinterpret_cast<__bf16*>(d_ws);

  // convert + transpose + scale all weights into d_ws (bf16, [N][K])
  for (int w = 0; w < NWT; ++w) {
    const float* src = (const float*)d_in[1 + w];
    int K = WKd[w], N = WNd[w];
    int tot = K * N;
    prep_w<<<(tot + 255) / 256, 256, 0, stream>>>(src, W + WOFF.v[w], K, N, WSCALE[w]);
  }

  equi_main<<<8192 / TILE_B, 256, LDS_BYTES, stream>>>(x, W, (float*)d_out);
}
